// MultiRoundLSHAttention_44856638439749
// MI455X (gfx1250) — matrix operand order repaired and checked by reading, not compiled, not run on hardware
//
#include <hip/hip_runtime.h>

// ---------------------------------------------------------------------------
// Multi-round LSH attention for MI455X (gfx1250, wave32, WMMA bf16)
// ---------------------------------------------------------------------------
#define B_   1
#define L_   4096
#define DM_  512
#define HEAD_ 8
#define DK_  64
#define R_   4
#define NB2_ 32
#define BL_  64
#define BH_  (B_*HEAD_)

typedef unsigned short u16;
typedef unsigned int   u32;
typedef __attribute__((ext_vector_type(16))) __bf16 bf16x16;
typedef __attribute__((ext_vector_type(8)))  float  f32x8;

union FragU { bf16x16 v; uint4 u[2]; };

static __device__ inline u16 f2bf(float f) {
    u32 u = __float_as_uint(f);
    u32 r = u + 0x7FFFu + ((u >> 16) & 1u);
    return (u16)(r >> 16);
}

// A fragment (16x32 bf16, M x K): lane holds row (lane&15), K half (lane>>4)
static __device__ inline bf16x16 ldfragA(const u16* p) {
    FragU f;
    f.u[0] = *(const uint4*)(p);
    f.u[1] = *(const uint4*)(p + 8);
    return f.v;
}
// B fragment (32x16 bf16, K x N): lane holds column (lane&15), K half (lane>>4)
static __device__ inline bf16x16 ldfragB(const u16* p) {
    FragU f;
    f.u[0] = *(const uint4*)(p);
    f.u[1] = *(const uint4*)(p + 8);
    return f.v;
}

static __device__ inline f32x8 wmma_bf16(bf16x16 a, bf16x16 b, f32x8 c) {
    return __builtin_amdgcn_wmma_f32_16x16x32_bf16(false, a, false, b,
                                                   (short)0, c, false, false);
}

// ---------------------------------------------------------------------------
// K0: fp32 -> bf16 conversion (one element per thread)
// ---------------------------------------------------------------------------
__global__ __launch_bounds__(256) void cvt_bf16_kernel(const float* __restrict__ s,
                                                       u16* __restrict__ d, int n) {
    int i = blockIdx.x * 256 + threadIdx.x;
    if (i < n) d[i] = f2bf(s[i]);
}

// ---------------------------------------------------------------------------
// K0b: normalize LSH projections; rmn[bh][r][kk][d] (contiguous in d)
// rand layout: [Bh][DK][R][NB2]
// ---------------------------------------------------------------------------
__global__ __launch_bounds__(256) void rmn_kernel(const float* __restrict__ rnd,
                                                  float* __restrict__ rmn) {
    int t = blockIdx.x * 256 + threadIdx.x;   // 0..1023
    if (t >= BH_ * R_ * NB2_) return;
    int kk = t & 31, r = (t >> 5) & 3, bh = t >> 7;
    float ss = 0.f;
    for (int d = 0; d < DK_; ++d) {
        float v = rnd[((size_t)(bh * DK_ + d) * R_ + r) * NB2_ + kk];
        ss += v * v;
    }
    float inv = 1.0f / sqrtf(ss);
    for (int d = 0; d < DK_; ++d) {
        float v = rnd[((size_t)(bh * DK_ + d) * R_ + r) * NB2_ + kk];
        rmn[((size_t)(bh * R_ + r) * NB2_ + kk) * DK_ + d] = v * inv;
    }
}

// ---------------------------------------------------------------------------
// K1/K6: C[M x 512] = A_bf16[M x 512] * W_bf16[512 x 512]^T + bias
// 128x128 tiles, 8 waves, wave owns a 16x128 strip; B-fragments are
// software-pipelined so ds_load_b128s overlap the matrix pipe.
// ---------------------------------------------------------------------------
__global__ __launch_bounds__(256) void gemm512_bf16(const u16* __restrict__ A,
                                                    const u16* __restrict__ W,
                                                    const float* __restrict__ bias,
                                                    float* __restrict__ C) {
    __shared__ u16 At[128 * 72];
    __shared__ u16 Wt[128 * 72];
    int tid = threadIdx.x;
    int bn = blockIdx.x & 3;            // 512/128 = 4 column blocks
    int bm = blockIdx.x >> 2;
    int w = tid >> 5, lane = tid & 31, l15 = lane & 15, kh = lane >> 4;

    f32x8 acc[8];
#pragma unroll
    for (int t = 0; t < 8; ++t) acc[t] = (f32x8){};

    for (int kc = 0; kc < 8; ++kc) {
        for (int t = tid; t < 1024; t += 256) {
            int row = t >> 3, c = t & 7;
            *(uint4*)&At[row * 72 + c * 8] =
                *(const uint4*)&A[(size_t)(bm * 128 + row) * 512 + kc * 64 + c * 8];
            *(uint4*)&Wt[row * 72 + c * 8] =
                *(const uint4*)&W[(size_t)(bn * 128 + row) * 512 + kc * 64 + c * 8];
        }
        if (kc < 7) {  // prefetch next A chunk (global_prefetch_b8)
            __builtin_prefetch(&A[(size_t)(bm * 128 + (tid >> 1)) * 512 + (kc + 1) * 64], 0, 0);
        }
        __syncthreads();
        bf16x16 a0 = ldfragA(&At[(w * 16 + l15) * 72 + 0 + kh * 16]);
        bf16x16 a1 = ldfragA(&At[(w * 16 + l15) * 72 + 32 + kh * 16]);
        bf16x16 b0 = ldfragB(&Wt[l15 * 72 + 0 + kh * 16]);
        bf16x16 b1 = ldfragB(&Wt[l15 * 72 + 32 + kh * 16]);
#pragma unroll
        for (int tn = 0; tn < 8; ++tn) {
            bf16x16 nb0, nb1;
            if (tn < 7) {   // prefetch next tile's fragments before the wmmas
                nb0 = ldfragB(&Wt[((tn + 1) * 16 + l15) * 72 + 0 + kh * 16]);
                nb1 = ldfragB(&Wt[((tn + 1) * 16 + l15) * 72 + 32 + kh * 16]);
            }
            acc[tn] = wmma_bf16(a0, b0, acc[tn]);
            acc[tn] = wmma_bf16(a1, b1, acc[tn]);
            b0 = nb0; b1 = nb1;
        }
        __syncthreads();
    }
#pragma unroll
    for (int tn = 0; tn < 8; ++tn) {
        int col = bn * 128 + tn * 16 + l15;
        float bv = bias[col];
#pragma unroll
        for (int e = 0; e < 8; ++e) {
            int row = bm * 128 + w * 16 + e + kh * 8;
            C[(size_t)row * 512 + col] = acc[tn][e] + bv;
        }
    }
}

// ---------------------------------------------------------------------------
// K2: split heads, L2-normalize q, convert to bf16, compute LSH hashes
// ---------------------------------------------------------------------------
__global__ __launch_bounds__(256) void headsplit_hash(const float* __restrict__ qproj,
                                                      const float* __restrict__ vproj,
                                                      const float* __restrict__ rmn,
                                                      u16* __restrict__ qh,
                                                      u16* __restrict__ vh,
                                                      u32* __restrict__ keys) {
    int g = blockIdx.x * 256 + threadIdx.x;   // (bh, l)
    int bh = g >> 12, l = g & 4095;
    const float* qrow = qproj + (size_t)l * DM_ + bh * DK_;
    float qn[DK_];
    float ss = 0.f;
#pragma unroll
    for (int d = 0; d < DK_; ++d) { qn[d] = qrow[d]; ss += qn[d] * qn[d]; }
    float inv = 1.0f / sqrtf(ss);
#pragma unroll
    for (int d = 0; d < DK_; ++d) {
        qn[d] *= inv;
        qh[((size_t)bh * L_ + l) * DK_ + d] = f2bf(qn[d]);
    }
    const float* vrow = vproj + (size_t)l * DM_ + bh * DK_;
#pragma unroll
    for (int d = 0; d < DK_; ++d)
        vh[((size_t)bh * L_ + l) * DK_ + d] = f2bf(vrow[d]);

    for (int r = 0; r < R_; ++r) {
        float x[NB2_];
        const float* rp = rmn + (size_t)(bh * R_ + r) * NB2_ * DK_;
        for (int kk = 0; kk < NB2_; ++kk) {
            float s = 0.f;
#pragma unroll
            for (int d = 0; d < DK_; ++d) s += qn[d] * rp[kk * DK_ + d];
            x[kk] = s;
        }
        float best = -1e30f; int bi = 0;
        for (int k = 0; k < 64; ++k) {
            float v = (k < NB2_) ? x[k] : -x[k - NB2_];
            if (v > best) { best = v; bi = k; }
        }
        keys[(size_t)(bh * R_ + r) * L_ + l] = ((u32)bi << 12) | (u32)l;
    }
}

// ---------------------------------------------------------------------------
// K3: stable sort per (bh, r): bitonic on composite (hash<<12 | idx)
// ---------------------------------------------------------------------------
__global__ __launch_bounds__(1024) void bitonic_sort(u32* __restrict__ keys,
                                                     u32* __restrict__ opos) {
    __shared__ u32 sb[L_];
    int tid = threadIdx.x;
    u32* base = keys + (size_t)blockIdx.x * L_;
    for (int i = tid; i < L_; i += 1024) sb[i] = base[i];
    __syncthreads();
    for (unsigned k = 2; k <= (unsigned)L_; k <<= 1) {
        for (unsigned j = k >> 1; j > 0; j >>= 1) {
            for (unsigned t = tid; t < L_ / 2; t += 1024) {
                unsigned i = ((t & ~(j - 1)) << 1) | (t & (j - 1));
                unsigned p = i | j;
                u32 a = sb[i], b = sb[p];
                bool asc = ((i & k) == 0);
                if ((a > b) == asc) { sb[i] = b; sb[p] = a; }
            }
            __syncthreads();
        }
    }
    u32* op = opos + (size_t)blockIdx.x * L_;
    for (int p = tid; p < L_; p += 1024) {
        u32 c = sb[p];
        base[p] = c;                 // sorted composites back in place
        op[c & 4095u] = (u32)p;      // inverse permutation
    }
}

// ---------------------------------------------------------------------------
// K3b: pack per-round bucket ids (pos>>7) into one u32 per (bh, l)
// ---------------------------------------------------------------------------
__global__ __launch_bounds__(256) void bpack_kernel(const u32* __restrict__ opos,
                                                    u32* __restrict__ bpk) {
    int g = blockIdx.x * 256 + threadIdx.x;
    int bh = g >> 12, l = g & 4095;
    u32 b = 0;
#pragma unroll
    for (int r = 0; r < R_; ++r)
        b |= ((opos[(size_t)(bh * R_ + r) * L_ + l] >> 7) & 255u) << (8 * r);
    bpk[(size_t)bh * L_ + l] = b;
}

// ---------------------------------------------------------------------------
// K4 phase A: per (bh, r, bucket): WMMA scores + masks + count, row softmax
// (local), WMMA PV, store (m, Z, O) partials per sorted position.
// LDS map (bytes):
//   qt   0      128x72 bf16 (18432)
//   kt   18432  256x72 bf16 (36864)
//   vT   55296   64x264 bf16 (33792)   value tile, transposed [d][j]
//   qm   89088  u32[128], km 89600 u32[256], bq 90624 u32[128], bk 91136 u32[256]
//   S    92160  8 waves x 16 x 257 f32 (131584)
//   P    223744 8 waves x 16 x 264 bf16 (67584)   total 291328
// ---------------------------------------------------------------------------
#define SMEM_A 291328

__global__ __launch_bounds__(256) void lsh_bucket_attn(const u16* __restrict__ qh,
                                                       const u16* __restrict__ vh,
                                                       const u32* __restrict__ sortedc,
                                                       const u32* __restrict__ bpk,
                                                       const unsigned char* __restrict__ maskin,
                                                       float* __restrict__ Mpart,
                                                       float* __restrict__ Zpart,
                                                       float* __restrict__ Opart) {
    extern __shared__ char smem[];
    u16* qt = (u16*)(smem);
    u16* kt = (u16*)(smem + 18432);
    u16* vT = (u16*)(smem + 55296);
    u32* qm = (u32*)(smem + 89088);
    u32* km = (u32*)(smem + 89600);
    u32* bq = (u32*)(smem + 90624);
    u32* bk = (u32*)(smem + 91136);
    float* S = (float*)(smem + 92160);
    u16* P = (u16*)(smem + 223744);

    int tid = threadIdx.x;
    int blk = blockIdx.x;
    int n = blk % NB2_;
    int r = (blk / NB2_) % R_;
    int bh = blk / (NB2_ * R_);
    const u32* sc = sortedc + (size_t)(bh * R_ + r) * L_;

    // ---- meta ----
    if (tid < 128) {
        u32 c = sc[n * 128 + tid];
        qm[tid] = c;
        bq[tid] = bpk[(size_t)bh * L_ + (c & 4095u)];
    }
    {
        int j = tid;  // 0..255
        u32 v = 0xFFFFFFFFu, bv = 0;
        if (!(n == 0 && j < 128)) {
            int pk = (j < 128) ? (n - 1) * 128 + j : n * 128 + (j - 128);
            u32 c = sc[pk];
            u32 id = c & 4095u;
            if (!maskin[id]) c |= 0x80000000u;
            v = c;
            bv = bpk[(size_t)bh * L_ + id];
        }
        km[j] = v; bk[j] = bv;
    }
    __syncthreads();

    // ---- stage Q rows (gathered, keys == normalized queries) ----
    for (int t = tid; t < 128 * 8; t += 256) {
        int row = t >> 3, c = t & 7;
        int id = qm[row] & 4095;
        *(uint4*)(qt + row * 72 + c * 8) =
            *(const uint4*)(qh + ((size_t)bh * L_ + id) * DK_ + c * 8);
    }
    // ---- stage K rows ----
    for (int t = tid; t < 256 * 8; t += 256) {
        int row = t >> 3, c = t & 7;
        u32 m = km[row];
        uint4 val = make_uint4(0, 0, 0, 0);
        if (m != 0xFFFFFFFFu) {
            int id = m & 4095;
            val = *(const uint4*)(qh + ((size_t)bh * L_ + id) * DK_ + c * 8);
        }
        *(uint4*)(kt + row * 72 + c * 8) = val;
    }
    // ---- stage V transposed [d][j] ----
    for (int t = tid; t < 256 * 8; t += 256) {
        int j = t >> 3, c = t & 7;
        u32 m = km[j];
        if (m == 0xFFFFFFFFu) {
#pragma unroll
            for (int u = 0; u < 8; ++u) vT[(c * 8 + u) * 264 + j] = 0;
        } else {
            int id = m & 4095;
            const u16* src = vh + ((size_t)bh * L_ + id) * DK_ + c * 8;
#pragma unroll
            for (int u = 0; u < 8; ++u) vT[(c * 8 + u) * 264 + j] = src[u];
        }
    }
    __syncthreads();

    int w = tid >> 5, lane = tid & 31, l15 = lane & 15, kh = lane >> 4;
    float* Sw = S + w * 16 * 257;
    u16* Pw = P + (size_t)w * 16 * 264;
    int rowBase = w * 16;

    // ---- scores: wave w owns rows [16w, 16w+16) x 256 cols ----
    // A fragments depend only on the K-step: hoist out of the j-tile loop.
    bf16x16 qa0 = ldfragA(qt + (rowBase + l15) * 72 + 0 + kh * 16);
    bf16x16 qa1 = ldfragA(qt + (rowBase + l15) * 72 + 32 + kh * 16);
    {
        bf16x16 b0 = ldfragB(kt + l15 * 72 + 0 + kh * 16);
        bf16x16 b1 = ldfragB(kt + l15 * 72 + 32 + kh * 16);
        for (int jt = 0; jt < 16; ++jt) {
            f32x8 acc = (f32x8){};
            acc = wmma_bf16(qa0, b0, acc);
            acc = wmma_bf16(qa1, b1, acc);
            if (jt < 15) {  // pipeline next tile's B fragments past the wmmas
                b0 = ldfragB(kt + ((jt + 1) * 16 + l15) * 72 + 0 + kh * 16);
                b1 = ldfragB(kt + ((jt + 1) * 16 + l15) * 72 + 32 + kh * 16);
            }
            int col = jt * 16 + l15;
            u32 kmv = km[col];
            u32 bkv = bk[col];
#pragma unroll
            for (int e = 0; e < 8; ++e) {
                int rloc = e + kh * 8;
                u32 qmv = qm[rowBase + rloc];
                u32 bqv = bq[rowBase + rloc];
                float s = acc[e] * 0.125f;   // 1/sqrt(64)
                if (kmv == 0xFFFFFFFFu) {
                    s = -1e9f;               // look-back pad
                } else {
                    u32 kid = kmv & 4095u, khash = (kmv >> 12) & 63u;
                    u32 qid = qmv & 4095u, qhash = (qmv >> 12) & 63u;
                    if (kmv & 0x80000000u) s = -1e9f;   // sequence mask
                    if (qhash != khash)    s = -1e9f;   // different hash bucket
                    if (qid < kid)         s = -1e9f;   // causal on original idx
                    if (qid == kid)        s = -1e5f;   // self (applied last)
                }
                int cnt = 0;
#pragma unroll
                for (int rr = 0; rr < 4; ++rr) {
                    u32 x = (bqv >> (8 * rr)) & 255u;
                    u32 y = (bkv >> (8 * rr)) & 255u;
                    cnt += (x == y) || (y + 1u == x);
                }
                if (cnt < 1) cnt = 1;
                s -= __logf((float)cnt);
                Sw[rloc * 257 + col] = s;
            }
        }
    }

    // ---- wave-local row softmax (2 lanes per row) ----
    int srow = lane >> 1, half = lane & 1;
    float mmax = -1e30f;
    for (int c = half * 128; c < half * 128 + 128; ++c)
        mmax = fmaxf(mmax, Sw[srow * 257 + c]);
    mmax = fmaxf(mmax, __shfl_xor(mmax, 1, 32));
    float Zs = 0.f;
    for (int c = half * 128; c < half * 128 + 128; ++c) {
        float e = __expf(Sw[srow * 257 + c] - mmax);
        Zs += e;
        Pw[srow * 264 + c] = f2bf(e);
    }
    Zs += __shfl_xor(Zs, 1, 32);
    if (half == 0) {
        int p = n * 128 + rowBase + srow;
        size_t idx = (size_t)(bh * R_ + r) * L_ + p;
        Mpart[idx] = mmax;
        Zpart[idx] = Zs;
    }

    // ---- PV: O(16x64) = P(16x256) x V(256x64) ----
    // ks outer so each P fragment is loaded once and feeds 4 wmmas.
    {
        f32x8 oacc[4];
#pragma unroll
        for (int dn = 0; dn < 4; ++dn) oacc[dn] = (f32x8){};
#pragma unroll
        for (int ks = 0; ks < 8; ++ks) {
            bf16x16 a = ldfragA(Pw + l15 * 264 + ks * 32 + kh * 16);
#pragma unroll
            for (int dn = 0; dn < 4; ++dn) {
                bf16x16 b = ldfragB(vT + (dn * 16 + l15) * 264 + ks * 32 + kh * 16);
                oacc[dn] = wmma_bf16(a, b, oacc[dn]);
            }
        }
#pragma unroll
        for (int dn = 0; dn < 4; ++dn) {
#pragma unroll
            for (int e = 0; e < 8; ++e) {
                int p = n * 128 + rowBase + e + kh * 8;
                int d = dn * 16 + l15;
                Opart[((size_t)(bh * R_ + r) * L_ + p) * DK_ + d] = oacc[dn][e];
            }
        }
    }
}

// ---------------------------------------------------------------------------
// K5 phase B: merge the 4 rounds per original query; emit att as bf16 [L][512]
// ---------------------------------------------------------------------------
__global__ __launch_bounds__(256) void merge_rounds(const float* __restrict__ Mp,
                                                    const float* __restrict__ Zp,
                                                    const float* __restrict__ Op,
                                                    const u32* __restrict__ opos,
                                                    u16* __restrict__ attb) {
    int g = blockIdx.x * 4 + (threadIdx.x >> 6);  // query index (bh,l)
    int d = threadIdx.x & 63;
    int bh = g >> 12, l = g & 4095;
    float mr[R_];
    u32 pr[R_];
    float M = -1e30f;
#pragma unroll
    for (int r = 0; r < R_; ++r) {
        size_t base = (size_t)(bh * R_ + r) * L_;
        pr[r] = opos[base + l];
        mr[r] = Mp[base + pr[r]];
        M = fmaxf(M, mr[r]);
    }
    float Z = 0.f, acc = 0.f;
#pragma unroll
    for (int r = 0; r < R_; ++r) {
        size_t idx = (size_t)(bh * R_ + r) * L_ + pr[r];
        float wgt = __expf(mr[r] - M);
        Z += wgt * Zp[idx];
        acc += wgt * Op[idx * DK_ + d];
    }
    attb[(size_t)l * DM_ + bh * DK_ + d] = f2bf(acc / Z);
}

// ---------------------------------------------------------------------------
// host launcher
// ---------------------------------------------------------------------------
extern "C" void kernel_launch(void* const* d_in, const int* in_sizes, int n_in,
                              void* d_out, int out_size, void* d_ws, size_t ws_size,
                              hipStream_t stream) {
    (void)in_sizes; (void)n_in; (void)out_size; (void)ws_size;
    const float* query = (const float*)d_in[0];
    const float* value = (const float*)d_in[1];
    const unsigned char* mask = (const unsigned char*)d_in[2];
    const float* Wq = (const float*)d_in[3];
    const float* bq = (const float*)d_in[4];
    const float* Wv = (const float*)d_in[5];
    const float* bv = (const float*)d_in[6];
    const float* Wo = (const float*)d_in[7];
    const float* bo = (const float*)d_in[8];
    const float* rnd = (const float*)d_in[9];

    char* ws = (char*)d_ws;
    size_t off = 0;
    auto carve = [&](size_t bytes) -> char* {
        char* p = ws + off;
        off = (off + bytes + 255) & ~(size_t)255;
        return p;
    };
    u16* wqb   = (u16*)carve((size_t)DM_ * DM_ * 2);
    u16* wvb   = (u16*)carve((size_t)DM_ * DM_ * 2);
    u16* wob   = (u16*)carve((size_t)DM_ * DM_ * 2);
    u16* qryb  = (u16*)carve((size_t)L_ * DM_ * 2);
    u16* valb  = (u16*)carve((size_t)L_ * DM_ * 2);
    u16* attb  = (u16*)carve((size_t)L_ * DM_ * 2);
    float* qproj = (float*)carve((size_t)L_ * DM_ * 4);
    float* vproj = (float*)carve((size_t)L_ * DM_ * 4);
    u16* qhb   = (u16*)carve((size_t)BH_ * L_ * DK_ * 2);
    u16* vhb   = (u16*)carve((size_t)BH_ * L_ * DK_ * 2);
    float* rmn = (float*)carve((size_t)BH_ * R_ * NB2_ * DK_ * 4);
    u32* keys  = (u32*)carve((size_t)BH_ * R_ * L_ * 4);
    u32* opos  = (u32*)carve((size_t)BH_ * R_ * L_ * 4);
    u32* bpk   = (u32*)carve((size_t)BH_ * L_ * 4);
    float* Mpart = (float*)carve((size_t)BH_ * R_ * L_ * 4);
    float* Zpart = (float*)carve((size_t)BH_ * R_ * L_ * 4);
    float* Opart = (float*)carve((size_t)BH_ * R_ * L_ * DK_ * 4);

    const int nLD = L_ * DM_, nWW = DM_ * DM_;
    cvt_bf16_kernel<<<(nLD + 255) / 256, 256, 0, stream>>>(query, qryb, nLD);
    cvt_bf16_kernel<<<(nLD + 255) / 256, 256, 0, stream>>>(value, valb, nLD);
    cvt_bf16_kernel<<<(nWW + 255) / 256, 256, 0, stream>>>(Wq, wqb, nWW);
    cvt_bf16_kernel<<<(nWW + 255) / 256, 256, 0, stream>>>(Wv, wvb, nWW);
    cvt_bf16_kernel<<<(nWW + 255) / 256, 256, 0, stream>>>(Wo, wob, nWW);
    rmn_kernel<<<4, 256, 0, stream>>>(rnd, rmn);

    // projections: q = query Wq^T + bq ; v = value Wv^T + bv
    gemm512_bf16<<<(L_ / 128) * 4, 256, 0, stream>>>(qryb, wqb, bq, qproj);
    gemm512_bf16<<<(L_ / 128) * 4, 256, 0, stream>>>(valb, wvb, bv, vproj);

    headsplit_hash<<<BH_ * L_ / 256, 256, 0, stream>>>(qproj, vproj, rmn, qhb, vhb, keys);
    bitonic_sort<<<BH_ * R_, 1024, 0, stream>>>(keys, opos);
    bpack_kernel<<<BH_ * L_ / 256, 256, 0, stream>>>(opos, bpk);

    lsh_bucket_attn<<<BH_ * R_ * NB2_, 256, SMEM_A, stream>>>(qhb, vhb, keys, bpk,
                                                              mask, Mpart, Zpart, Opart);
    merge_rounds<<<BH_ * L_ / 4, 256, 0, stream>>>(Mpart, Zpart, Opart, opos, attb);

    // out = att Wo^T + bo
    gemm512_bf16<<<(L_ / 128) * 4, 256, 0, stream>>>(attb, wob, bo, (float*)d_out);
}
